// VanillaQuantize_62586263437527
// MI455X (gfx1250) — compile-verified, matrix operand order checked by reading, not run on hardware
//
#include <hip/hip_runtime.h>
#include <hip/hip_bf16.h>

// ---------------------------------------------------------------------------
// VQ-VAE vector quantize for MI455X (gfx1250, wave32, WMMA)
//   x            [32,64,64,64] f32 (NCHW)   -> N = 131072 rows of dim 64
//   dictionary   [64,512] f32
//   cluster_size [512] f32
//   dictionary_avg [64,512] f32
// Outputs (flat, f32, in reference return order):
//   out[8388608], diff[1], embed_ind[131072], new_cluster_size[512],
//   new_dictionary_avg[32768], new_dictionary[32768], mean_D[1]
// ---------------------------------------------------------------------------

#define VQ_DIM      64
#define VQ_NEMBED   512
#define VQ_HW       4096       // 64*64
#define VQ_NROWS    131072     // 32*64*64
#define VQ_DECAY    0.99f
#define VQ_ALPHA    0.01f
#define VQ_EPS      1e-5f

typedef __attribute__((ext_vector_type(16))) __bf16 v16bf;
typedef __attribute__((ext_vector_type(8)))  float  v8f;

union BFrag {
  v16bf v;
  unsigned short u[16];
  uint4 q[2];
};

static __device__ __forceinline__ unsigned short f32_to_bf16_rne(float f) {
  unsigned b = __builtin_bit_cast(unsigned, f);
  b += 0x7FFFu + ((b >> 16) & 1u);           // round-to-nearest-even
  return (unsigned short)(b >> 16);
}

// K index held by element e of a 16-bit A/B WMMA fragment (ISA 7.12.2):
// lanes 0-15 hold K in {0..7, 16..23}; lanes 16-31 hold K in {8..15, 24..31}.
static __device__ __forceinline__ int kmap(int e, bool hiHalf) {
  return (e < 8 ? e : e + 8) + (hiHalf ? 8 : 0);
}

// Async global->LDS staging path (ASYNCcnt), with safe fallback.
#if defined(__HIP_DEVICE_COMPILE__) && \
    __has_builtin(__builtin_amdgcn_global_load_async_to_lds_b128) && \
    __has_builtin(__builtin_amdgcn_s_wait_asynccnt)
#define VQ_ASYNC_LDS 1
#else
#define VQ_ASYNC_LDS 0
#endif

// ---------------------------------------------------------------------------
// Kernel 1: pack dictionary into wave32 WMMA B-fragment layout (bf16),
// compute ||D_j||^2, zero the accumulators. 32768 threads.
// ws_b ushort layout (16B-chunked for ds_load_b128):
//   idx = ((t*2+kb)*2 + half)*256 + lane*8 + e8,  fragment element e = half*8+e8
// ---------------------------------------------------------------------------
__global__ __launch_bounds__(256)
void vq_prep_kernel(const float* __restrict__ dict,
                    unsigned short* __restrict__ ws_b,
                    float* __restrict__ ws_normD,
                    float* __restrict__ ws_counts,
                    float* __restrict__ ws_esum,
                    float* __restrict__ ws_diff) {
  int idx = blockIdx.x * 256 + threadIdx.x;          // 0..32767
  int e8   = idx & 7;
  int lane = (idx >> 3) & 31;
  int half = (idx >> 8) & 1;
  int tk   = idx >> 9;                               // t*2 + kbSel, 0..63
  int t    = tk >> 1;
  int kb   = (tk & 1) * 32;
  int N    = t * 16 + (lane & 15);
  int K    = kb + kmap(half * 8 + e8, lane >= 16);
  ws_b[idx] = f32_to_bf16_rne(dict[K * VQ_NEMBED + N]);

  ws_esum[idx] = 0.0f;                               // zero [64,512] accum

  if (idx < VQ_NEMBED) {
    float s = 0.0f;
    #pragma unroll 8
    for (int c = 0; c < VQ_DIM; ++c) {
      float d = dict[c * VQ_NEMBED + idx];
      s += d * d;
    }
    ws_normD[idx]  = s;
    ws_counts[idx] = 0.0f;
  }
  if (idx == 0) ws_diff[0] = 0.0f;
}

// ---------------------------------------------------------------------------
// Kernel 2: WMMA nearest-code search + quantize write + scatter accumulate.
// 1024 blocks x 256 threads (8 waves); each wave owns 16 rows x all 512 codes.
// B operand staged once per block into LDS (async global->LDS if available).
// ---------------------------------------------------------------------------
__global__ __launch_bounds__(256)
void vq_argmin_kernel(const float* __restrict__ x,
                      const float* __restrict__ dict,
                      const unsigned short* __restrict__ ws_b,
                      const float* __restrict__ ws_normD,
                      float* __restrict__ ws_counts,
                      float* __restrict__ ws_esum,
                      float* __restrict__ ws_diff,
                      float* __restrict__ out_q,
                      float* __restrict__ out_ind) {
  __shared__ uint4 s_b[4096];          // 64 KB packed bf16 B fragments
  __shared__ float s_normD[VQ_NEMBED]; // 2 KB
  __shared__ int   s_inds[128];

  // ---- stage B into LDS (ASYNCcnt path on CDNA5) ----
#if VQ_ASYNC_LDS
  {
    typedef int v4i_t __attribute__((vector_size(16)));
    typedef __attribute__((address_space(1))) v4i_t gv4i_t;
    typedef __attribute__((address_space(3))) v4i_t lv4i_t;
    gv4i_t* g = (gv4i_t*)ws_b;           // global src, 16B elements
    lv4i_t* l = (lv4i_t*)s_b;            // LDS dst, 16B elements
    for (int i = threadIdx.x; i < 4096; i += 256)
      __builtin_amdgcn_global_load_async_to_lds_b128(g + i, l + i, 0, 0);
    __builtin_amdgcn_s_wait_asynccnt(0);   // own wave's async copies done
  }
#else
  {
    uint4* dstq = (uint4*)s_b;
    const uint4* srcq = (const uint4*)ws_b;
    for (int i = threadIdx.x; i < 4096; i += 256) dstq[i] = srcq[i];
  }
#endif
  for (int i = threadIdx.x; i < VQ_NEMBED; i += 256)
    s_normD[i] = ws_normD[i];
  __syncthreads();

  const int  lane    = threadIdx.x & 31;
  const int  wave    = threadIdx.x >> 5;
  const bool hiHalf  = lane >= 16;
  const int  rowBase = blockIdx.x * 128 + wave * 16;

  // A fragments: row M = lane%16 of the 16x64 tile, gathered from NCHW.
  const int myRow = rowBase + (lane & 15);
  const int w0 = myRow & 63, h0 = (myRow >> 6) & 63, b0 = myRow >> 12;
  const float* xrow = x + (size_t)b0 * VQ_DIM * VQ_HW + h0 * 64 + w0;

  BFrag a0, a1;                                      // K = 0..31, 32..63
  #pragma unroll
  for (int e = 0; e < 16; ++e) {
    int k = kmap(e, hiHalf);
    a0.u[e] = f32_to_bf16_rne(xrow[(size_t)k * VQ_HW]);
    a1.u[e] = f32_to_bf16_rne(xrow[(size_t)(k + 32) * VQ_HW]);
  }

  // Incremental argmin of (||D||^2 - 2 f.D): 8 C-matrix slots per lane.
  float best[8];
  int   bidx[8];
  #pragma unroll
  for (int v = 0; v < 8; ++v) { best[v] = 3.402823466e38f; bidx[v] = 0; }

  const uint4* sbq = (const uint4*)s_b;
  // Two j-tiles per iteration with independent accumulators: the two extra
  // independent WMMAs cover the WMMA->VALU read hazard (ISA 7.12.1).
  for (int t = 0; t < 32; t += 2) {
    const int base = t * 128;                        // uint4 chunks
    BFrag bA0, bA1, bB0, bB1;
    bA0.q[0] = sbq[base +       lane];
    bA0.q[1] = sbq[base +  32 + lane];
    bA1.q[0] = sbq[base +  64 + lane];
    bA1.q[1] = sbq[base +  96 + lane];
    bB0.q[0] = sbq[base + 128 + lane];
    bB0.q[1] = sbq[base + 160 + lane];
    bB1.q[0] = sbq[base + 192 + lane];
    bB1.q[1] = sbq[base + 224 + lane];

    v8f accA = {};
    v8f accB = {};
    accA = __builtin_amdgcn_wmma_f32_16x16x32_bf16(false, a0.v, false, bA0.v,
                                                   (short)0, accA, false, false);
    accA = __builtin_amdgcn_wmma_f32_16x16x32_bf16(false, a1.v, false, bA1.v,
                                                   (short)0, accA, false, false);
    accB = __builtin_amdgcn_wmma_f32_16x16x32_bf16(false, a0.v, false, bB0.v,
                                                   (short)0, accB, false, false);
    accB = __builtin_amdgcn_wmma_f32_16x16x32_bf16(false, a1.v, false, bB1.v,
                                                   (short)0, accB, false, false);

    const int   jA = t * 16 + (lane & 15);
    const int   jB = jA + 16;
    const float nA = s_normD[jA];
    const float nB = s_normD[jB];
    #pragma unroll
    for (int v = 0; v < 8; ++v) {
      float sA = nA - 2.0f * accA[v];
      if (sA < best[v]) { best[v] = sA; bidx[v] = jA; }  // strict < keeps lowest j
    }
    #pragma unroll
    for (int v = 0; v < 8; ++v) {
      float sB = nB - 2.0f * accB[v];
      if (sB < best[v]) { best[v] = sB; bidx[v] = jB; }
    }
  }

  // Cross-lane min-reduce within each 16-lane half (columns of one row set).
  #pragma unroll
  for (int v = 0; v < 8; ++v) {
    #pragma unroll
    for (int off = 8; off >= 1; off >>= 1) {
      float os = __shfl_xor(best[v], off, 32);
      int   oi = __shfl_xor(bidx[v], off, 32);
      if (os < best[v] || (os == best[v] && oi < bidx[v])) {
        best[v] = os; bidx[v] = oi;
      }
    }
  }

  // Rows 0..7 live in lanes 0-15 (slots v), rows 8..15 in lanes 16-31.
  if ((lane & 15) == 0) {
    #pragma unroll
    for (int v = 0; v < 8; ++v)
      s_inds[wave * 16 + v + (hiHalf ? 8 : 0)] = bidx[v];
  }
  __syncthreads();

  // Quantize write-back (NCHW, coalesced over w), diff, and scatter updates.
  float diffAcc = 0.0f;
  for (int idx = lane; idx < 16 * VQ_DIM; idx += 32) {
    int c = idx >> 4;
    int r = idx & 15;
    int n = rowBase + r;
    int j = s_inds[wave * 16 + r];
    float dval = dict[c * VQ_NEMBED + j];
    int w2 = n & 63, h2 = (n >> 6) & 63, b2 = n >> 12;
    size_t off = (size_t)(b2 * VQ_DIM + c) * VQ_HW + h2 * 64 + w2;
    float xv = x[off];
    out_q[off] = dval;
    float d = dval - xv;
    diffAcc += d * d;
    atomicAdd(&ws_esum[c * VQ_NEMBED + j], xv);      // embed_sum scatter
  }
  if (lane < 16) {
    int j = s_inds[wave * 16 + lane];
    atomicAdd(&ws_counts[j], 1.0f);
    out_ind[rowBase + lane] = (float)j;
  }
  #pragma unroll
  for (int off = 16; off >= 1; off >>= 1)
    diffAcc += __shfl_xor(diffAcc, off, 32);
  if (lane == 0) atomicAdd(ws_diff, diffAcc);
}

// ---------------------------------------------------------------------------
// Kernel 3: EMA buffer math + mean_D + diff. One block, 512 threads.
// ---------------------------------------------------------------------------
__global__ __launch_bounds__(512)
void vq_finalize_kernel(const float* __restrict__ cluster_size,
                        const float* __restrict__ davg,
                        const float* __restrict__ ws_counts,
                        const float* __restrict__ ws_esum,
                        const float* __restrict__ ws_diff,
                        float* __restrict__ out_diff,
                        float* __restrict__ out_ncs,
                        float* __restrict__ out_nda,
                        float* __restrict__ out_nd,
                        float* __restrict__ out_meanD) {
  __shared__ float red[VQ_NEMBED];
  __shared__ float s_n;

  int j = threadIdx.x;
  float ncs = VQ_DECAY * cluster_size[j] + VQ_ALPHA * ws_counts[j];
  out_ncs[j] = ncs;

  red[j] = ncs;
  __syncthreads();
  #pragma unroll
  for (int s = 256; s >= 1; s >>= 1) {
    if (j < s) red[j] += red[j + s];
    __syncthreads();
  }
  if (j == 0) s_n = red[0];
  __syncthreads();
  float n  = s_n;
  float cs = (ncs + VQ_EPS) / (n + VQ_NEMBED * VQ_EPS) * n;

  float absAcc = 0.0f;
  #pragma unroll 4
  for (int c = 0; c < VQ_DIM; ++c) {
    float nda = VQ_DECAY * davg[c * VQ_NEMBED + j] +
                VQ_ALPHA * ws_esum[c * VQ_NEMBED + j];
    out_nda[c * VQ_NEMBED + j] = nda;
    float nd = nda / cs;
    out_nd[c * VQ_NEMBED + j] = nd;
    absAcc += fabsf(nd);
  }

  red[j] = absAcc;
  __syncthreads();
  #pragma unroll
  for (int s = 256; s >= 1; s >>= 1) {
    if (j < s) red[j] += red[j + s];
    __syncthreads();
  }
  if (j == 0) {
    out_meanD[0] = red[0] * (1.0f / (VQ_DIM * VQ_NEMBED));
    out_diff[0]  = ws_diff[0] * (1.0f / ((float)VQ_NROWS * (float)VQ_DIM));
  }
}

// ---------------------------------------------------------------------------
// Launch
// ---------------------------------------------------------------------------
extern "C" void kernel_launch(void* const* d_in, const int* in_sizes, int n_in,
                              void* d_out, int out_size, void* d_ws, size_t ws_size,
                              hipStream_t stream) {
  const float* x            = (const float*)d_in[0];
  const float* dict         = (const float*)d_in[1];
  const float* cluster_size = (const float*)d_in[2];
  const float* davg         = (const float*)d_in[3];

  float* out = (float*)d_out;
  // output section offsets (floats), reference return order
  const size_t OFF_OUT   = 0;              // 8388608
  const size_t OFF_DIFF  = 8388608;        // 1
  const size_t OFF_IND   = 8388609;        // 131072
  const size_t OFF_NCS   = 8519681;        // 512
  const size_t OFF_NDA   = 8520193;        // 32768
  const size_t OFF_ND    = 8552961;        // 32768
  const size_t OFF_MEAND = 8585729;        // 1

  char* ws = (char*)d_ws;
  unsigned short* ws_b  = (unsigned short*)(ws);            // 64 KB packed B
  float* ws_normD       = (float*)(ws + 65536);             // 2 KB
  float* ws_counts      = (float*)(ws + 67584);             // 2 KB
  float* ws_esum        = (float*)(ws + 69632);             // 128 KB
  float* ws_diff        = (float*)(ws + 200704);            // 4 B

  vq_prep_kernel<<<128, 256, 0, stream>>>(dict, ws_b, ws_normD, ws_counts,
                                          ws_esum, ws_diff);

  vq_argmin_kernel<<<1024, 256, 0, stream>>>(x, dict, ws_b, ws_normD,
                                             ws_counts, ws_esum, ws_diff,
                                             out + OFF_OUT, out + OFF_IND);

  vq_finalize_kernel<<<1, 512, 0, stream>>>(cluster_size, davg, ws_counts,
                                            ws_esum, ws_diff,
                                            out + OFF_DIFF, out + OFF_NCS,
                                            out + OFF_NDA, out + OFF_ND,
                                            out + OFF_MEAND);
}
